// SoftAttention_31714038513809
// MI455X (gfx1250) — matrix operand order repaired and checked by reading, not compiled, not run on hardware
//
#include <hip/hip_runtime.h>

typedef _Float16 half_t;
typedef __attribute__((ext_vector_type(16))) _Float16 v16h;
typedef __attribute__((ext_vector_type(8)))  _Float16 v8h;
typedef __attribute__((ext_vector_type(8)))  float    v8f;

#define BATCH 8
#define SEQ   2048
#define DIM   512

__device__ __forceinline__ v8f wmma_fp16(v16h a, v16h b, v8f c) {
  // D = A(16x32 f16) * B(32x16 f16) + C(16x16 f32)
  return __builtin_amdgcn_wmma_f32_16x16x32_f16(false, a, false, b, (short)0, c,
                                                false, false);
}

// A-operand fragment (16x32 f16) from a row-major row pointer.
// lane L: row m = L&15 ; element e -> k = kbase + 16*(L>>4) + e
// => two contiguous 16B segments at kbase+16h and kbase+16h+8.
__device__ __forceinline__ v16h load_a_frag(const half_t* rowptr, int kbase, int h) {
  const half_t* p = rowptr + kbase + 16 * h;
  v8h lo = *(const v8h*)(p);
  v8h hi = *(const v8h*)(p + 8);
  v16h r;
#pragma unroll
  for (int i = 0; i < 8; i++) { r[i] = lo[i]; r[i + 8] = hi[i]; }
  return r;
}

// ---------------------------------------------------------------- converts
__global__ __launch_bounds__(256) void cvt_x_kernel(const float* __restrict__ x,
                                                    half_t* __restrict__ xh) {
  size_t i = (size_t)blockIdx.x * 256 + threadIdx.x;
  xh[i] = (half_t)x[i];
}

// Wt[n][k] = W[k][n], f16 (so GEMM B-operand reads are k-contiguous per lane)
__global__ __launch_bounds__(256) void tr_w_kernel(
    const float* __restrict__ Wq, const float* __restrict__ Wk,
    const float* __restrict__ Wv, half_t* __restrict__ Wtq,
    half_t* __restrict__ Wtk, half_t* __restrict__ Wtv) {
  const float* W = blockIdx.z == 0 ? Wq : (blockIdx.z == 1 ? Wk : Wv);
  half_t* Wt = blockIdx.z == 0 ? Wtq : (blockIdx.z == 1 ? Wtk : Wtv);
  int i = blockIdx.x * 256 + threadIdx.x;   // 0 .. 512*512-1, exact
  int n = i >> 9, k = i & 511;
  Wt[i] = (half_t)W[k * 512 + n];
}

// ------------------------------------------------------- projection GEMMs
// grid (M/16, DIM/256, 3) ; block 128 (4 waves). Wave w owns 64 output cols.
// z=0 -> Q row-major, z=1 -> K row-major, z=2 -> V stored transposed [b][d][n].
__global__ __launch_bounds__(128) void proj_kernel(
    const half_t* __restrict__ xh, const half_t* __restrict__ Wtq,
    const half_t* __restrict__ Wtk, const half_t* __restrict__ Wtv,
    const float* __restrict__ bq, const float* __restrict__ bk,
    const float* __restrict__ bv, half_t* __restrict__ Qh,
    half_t* __restrict__ Kh, half_t* __restrict__ Vt) {
  const int lane = threadIdx.x & 31;
  const int wave = threadIdx.x >> 5;
  const int h = lane >> 4;
  const int ln = lane & 15;
  const int z = blockIdx.z;
  const half_t* Wt = z == 0 ? Wtq : (z == 1 ? Wtk : Wtv);
  const float* bias = z == 0 ? bq : (z == 1 ? bk : bv);
  const int mbase = blockIdx.x * 16;
  const int colbase = blockIdx.y * 256 + wave * 64;

  v8f C[4];
#pragma unroll
  for (int t = 0; t < 4; t++) C[t] = (v8f){0, 0, 0, 0, 0, 0, 0, 0};

  const half_t* xrow = xh + (size_t)(mbase + ln) * DIM;
  for (int kk = 0; kk < DIM; kk += 32) {
    v16h a = load_a_frag(xrow, kk, h);
#pragma unroll
    for (int t = 0; t < 4; t++) {
      // B-operand: b[k][n] = W[k][n] = Wt[n][k], k contiguous per lane
      const half_t* wrow = Wt + (size_t)(colbase + t * 16 + ln) * DIM + kk + 16 * h;
      v16h b = *(const v16h*)wrow;
      C[t] = wmma_fp16(a, b, C[t]);
    }
  }

  if (z < 2) {
    half_t* O = (z == 0) ? Qh : Kh;
#pragma unroll
    for (int t = 0; t < 4; t++) {
      int col = colbase + t * 16 + ln;
      float bb = bias[col];
#pragma unroll
      for (int r = 0; r < 8; r++)
        O[(size_t)(mbase + r + 8 * h) * DIM + col] = (half_t)(C[t][r] + bb);
    }
  } else {
    // transposed V: Vt[b][col][row] ; lane's 8 rows are consecutive -> 16B store
    int bidx = mbase >> 11;                 // /2048 (tiles never straddle batch)
    int rowInB = (mbase & 2047) + 8 * h;
#pragma unroll
    for (int t = 0; t < 4; t++) {
      int col = colbase + t * 16 + ln;
      float bb = bias[col];
      v8h pk;
#pragma unroll
      for (int r = 0; r < 8; r++) pk[r] = (half_t)(C[t][r] + bb);
      *(v8h*)(Vt + ((size_t)bidx * DIM + col) * SEQ + rowInB) = pk;
    }
  }
}

// ------------------------------------------------------ flash attention
// grid (SEQ/32, BATCH) ; block 128 (4 waves). 32 queries per block
// (halves K/V L2 traffic vs 16, doubles K-fragment reuse).
// KV tile = 64 keys: wave w computes S[:, 16w:16w+16] for both query
// row-tiles; then waves split the 512 output dims 128 each for O += P @ V.
#define QROWS  32
#define QPITCH 520   // Q rows in LDS, padded (bank-conflict free)
#define SPITCH 68
#define PPITCH 72

__global__ __launch_bounds__(128, 1) void attn_kernel(
    const half_t* __restrict__ Qh, const half_t* __restrict__ Kh,
    const half_t* __restrict__ Vt, const float* __restrict__ adj,
    const float* __restrict__ p_scale, const float* __restrict__ p_width,
    float* __restrict__ out) {
  const int lane = threadIdx.x & 31;
  const int wave = threadIdx.x >> 5;
  const int h = lane >> 4;
  const int ln = lane & 15;
  const int b = blockIdx.y;
  const int qbase = blockIdx.x * QROWS;

  __shared__ half_t Qs[QROWS * QPITCH];
  __shared__ float Stile[QROWS][SPITCH];
  __shared__ half_t Ptile[QROWS][PPITCH];
  __shared__ float smem_m[QROWS];
  __shared__ float smem_alpha[QROWS];
  __shared__ float smem_l[QROWS];

  const half_t* Qb = Qh + ((size_t)b * SEQ + qbase) * DIM;
  const half_t* Kb = Kh + (size_t)b * SEQ * DIM;
  const half_t* Vb = Vt + (size_t)b * DIM * SEQ;
  const float* adjb = adj + ((size_t)b * SEQ + qbase) * SEQ;

  const float inv_sqrt_d = 0.044194173824159216f;  // 1/sqrt(512)
  const float sc = *p_scale;
  const float inv_wd = 1.0f / (*p_width);

  // stage the 32x512 Q tile once into padded LDS
  for (int i = threadIdx.x; i < QROWS * 32; i += 128) {
    int r = i >> 5, sg = i & 31;
    const v8h* src = (const v8h*)(Qb + (size_t)r * DIM + sg * 16);
    v8h a0 = src[0];
    v8h a1 = src[1];
    v8h* dst = (v8h*)(&Qs[r * QPITCH + sg * 16]);
    dst[0] = a0;
    dst[1] = a1;
  }
  __syncthreads();

  v8f O[16];
#pragma unroll
  for (int t = 0; t < 16; t++) O[t] = (v8f){0, 0, 0, 0, 0, 0, 0, 0};

  // distributed softmax state: thread owns row srow (4 lanes per row)
  const int srow = 8 * wave + (lane >> 2);
  const int sseg = (lane & 3) * 16;
  float m_run = -INFINITY;
  float l_run = 0.0f;

  // adj prefetch pattern: thread covers one 64B chunk of next tile's mask
  const int pf_row = threadIdx.x >> 2;
  const int pf_col = (threadIdx.x & 3) * 16;

  for (int kv = 0; kv < SEQ; kv += 64) {
    // prefetch next KV tile's adj chunk into cache (global_prefetch_b8)
    {
      int kvn = (kv + 64 < SEQ) ? kv + 64 : 0;
      __builtin_prefetch(&adjb[(size_t)pf_row * SEQ + kvn + pf_col], 0, 1);
    }

    const int kcol = kv + wave * 16 + ln;

    // hoisted mask loads: retire underneath the QK^T WMMAs
    float adjv[16];
#pragma unroll
    for (int q = 0; q < 2; q++)
#pragma unroll
      for (int r = 0; r < 8; r++)
        adjv[q * 8 + r] = adjb[(size_t)(q * 16 + r + 8 * h) * SEQ + kcol];

    // ---- phase 1: S = Q @ K^T for this wave's 16 keys, both row-tiles
    v8f S[2];
    S[0] = (v8f){0, 0, 0, 0, 0, 0, 0, 0};
    S[1] = (v8f){0, 0, 0, 0, 0, 0, 0, 0};
    const half_t* Krow = Kb + (size_t)kcol * DIM;   // B-op: k-dim contiguous
    for (int kk = 0; kk < DIM; kk += 32) {
      v16h bb = *(const v16h*)(Krow + kk + 16 * h);
#pragma unroll
      for (int q = 0; q < 2; q++) {
        v16h a = load_a_frag(&Qs[(q * 16 + ln) * QPITCH], kk, h);
        S[q] = wmma_fp16(a, bb, S[q]);
      }
    }

    // ---- phase 2: fuse Gaussian mask, stash to LDS for cross-wave stats
#pragma unroll
    for (int q = 0; q < 2; q++) {
#pragma unroll
      for (int r = 0; r < 8; r++) {
        int row = q * 16 + r + 8 * h;
        float d = adjv[q * 8 + r] - sc;
        float mask = __expf(-(d * d) * inv_wd);
        float s = S[q][r] * inv_sqrt_d * mask;
        S[q][r] = s;
        Stile[row][wave * 16 + ln] = s;
      }
    }
    __syncthreads();

    // ---- phase 3: distributed per-row running max + alpha (4 lanes/row)
    float tm = Stile[srow][sseg];
#pragma unroll
    for (int j = 1; j < 16; j++) tm = fmaxf(tm, Stile[srow][sseg + j]);
    tm = fmaxf(tm, __shfl_xor(tm, 1, 32));
    tm = fmaxf(tm, __shfl_xor(tm, 2, 32));
    float mnew = fmaxf(m_run, tm);
    float alpha = __expf(m_run - mnew);
    m_run = mnew;
    if ((lane & 3) == 0) {
      smem_m[srow] = mnew;
      smem_alpha[srow] = alpha;
    }
    __syncthreads();

    // ---- phase 4: distributed exp -> P (f16 in LDS); rescale O by alpha
#pragma unroll
    for (int q = 0; q < 2; q++) {
#pragma unroll
      for (int r = 0; r < 8; r++) {
        int row = q * 16 + r + 8 * h;
        float p = __expf(S[q][r] - smem_m[row]);
        Ptile[row][wave * 16 + ln] = (half_t)p;
      }
    }
#pragma unroll
    for (int q = 0; q < 2; q++) {
#pragma unroll
      for (int t = 0; t < 8; t++) {
#pragma unroll
        for (int r = 0; r < 8; r++)
          O[q * 8 + t][r] *= smem_alpha[q * 16 + r + 8 * h];
      }
    }
    __syncthreads();

    // running denominator from the just-published P tile (distributed)
    {
      float psum = 0.0f;
#pragma unroll
      for (int j = 0; j < 16; j++) psum += (float)Ptile[srow][sseg + j];
      psum += __shfl_xor(psum, 1, 32);
      psum += __shfl_xor(psum, 2, 32);
      l_run = l_run * alpha + psum;
    }

    // ---- phase 5: O += P @ V  (V is pre-transposed: key-dim contiguous)
#pragma unroll
    for (int step = 0; step < 2; step++) {
      v16h aP[2];
#pragma unroll
      for (int q = 0; q < 2; q++)
        aP[q] = load_a_frag(&Ptile[q * 16 + ln][0], 32 * step, h);
#pragma unroll
      for (int t = 0; t < 8; t++) {
        int col = wave * 128 + t * 16 + ln;
        const half_t* Vrow = Vb + (size_t)col * SEQ + kv;
        v16h bb = *(const v16h*)(Vrow + 32 * step + 16 * h);
#pragma unroll
        for (int q = 0; q < 2; q++) O[q * 8 + t] = wmma_fp16(aP[q], bb, O[q * 8 + t]);
      }
    }
  }

  // ---- epilogue: divide by softmax denominator, store f32
  if ((lane & 3) == 0) smem_l[srow] = l_run;
  __syncthreads();
  float invl[16];
#pragma unroll
  for (int q = 0; q < 2; q++)
#pragma unroll
    for (int r = 0; r < 8; r++)
      invl[q * 8 + r] = 1.0f / smem_l[q * 16 + r + 8 * h];
#pragma unroll
  for (int q = 0; q < 2; q++) {
#pragma unroll
    for (int t = 0; t < 8; t++) {
      int col = wave * 128 + t * 16 + ln;
#pragma unroll
      for (int r = 0; r < 8; r++) {
        out[((size_t)b * SEQ + qbase + q * 16 + r + 8 * h) * DIM + col] =
            O[q * 8 + t][r] * invl[q * 8 + r];
      }
    }
  }
}

// ------------------------------------------------------------------ host
extern "C" void kernel_launch(void* const* d_in, const int* in_sizes, int n_in,
                              void* d_out, int out_size, void* d_ws,
                              size_t ws_size, hipStream_t stream) {
  const float* x   = (const float*)d_in[0];
  const float* adj = (const float*)d_in[1];
  const float* Wq  = (const float*)d_in[2];
  const float* bq  = (const float*)d_in[3];
  const float* Wk  = (const float*)d_in[4];
  const float* bk  = (const float*)d_in[5];
  const float* Wv  = (const float*)d_in[6];
  const float* bv  = (const float*)d_in[7];
  const float* scl = (const float*)d_in[8];
  const float* wid = (const float*)d_in[9];
  float* out = (float*)d_out;

  const size_t M = (size_t)BATCH * SEQ;         // 16384
  char* ws = (char*)d_ws;
  half_t* xh  = (half_t*)ws; ws += M * DIM * sizeof(half_t);
  half_t* Wtq = (half_t*)ws; ws += (size_t)DIM * DIM * sizeof(half_t);
  half_t* Wtk = (half_t*)ws; ws += (size_t)DIM * DIM * sizeof(half_t);
  half_t* Wtv = (half_t*)ws; ws += (size_t)DIM * DIM * sizeof(half_t);
  half_t* Qh  = (half_t*)ws; ws += M * DIM * sizeof(half_t);
  half_t* Kh  = (half_t*)ws; ws += M * DIM * sizeof(half_t);
  half_t* Vt  = (half_t*)ws; ws += M * DIM * sizeof(half_t);

  cvt_x_kernel<<<(unsigned)(M * DIM / 256), 256, 0, stream>>>(x, xh);
  tr_w_kernel<<<dim3(DIM * DIM / 256, 1, 3), 256, 0, stream>>>(Wq, Wk, Wv, Wtq,
                                                               Wtk, Wtv);
  proj_kernel<<<dim3((unsigned)(M / 16), DIM / 256, 3), 128, 0, stream>>>(
      xh, Wtq, Wtk, Wtv, bq, bk, bv, Qh, Kh, Vt);
  attn_kernel<<<dim3(SEQ / QROWS, BATCH), 128, 0, stream>>>(Qh, Kh, Vt, adj,
                                                            scl, wid, out);
}
